// GCN_JKNet2_27152783245354
// MI455X (gfx1250) — matrix operand order repaired and checked by reading, not compiled, not run on hardware
//
#include <hip/hip_runtime.h>
#include <hip/hip_bf16.h>
#include <math.h>

typedef _Float16 half_t;
typedef __attribute__((ext_vector_type(16))) _Float16 v16h;
typedef __attribute__((ext_vector_type(8)))  _Float16 v8h;
typedef __attribute__((ext_vector_type(8)))  float    v8f;

#define NFEAT 500
#define KPAD  512
#define NHID  256
#define NCLS  40

#define BM 64
#define BN 64
#define BK 32
#define LDSK 40   // padded LDS row stride (halves): 80B -> conflict-free fragment reads

// ---------------------------------------------------------------------------
// WMMA GEMM: C[M,Nout] = A[M,K](f16) * Bt[Nout,K](f16)^T  (+C if accFlag)
// Requires K % 32 == 0 and rows 16B-aligned. M/Nout arbitrary: out-of-range
// rows are CLAMPED on load (their products only reach discarded outputs) and
// guarded on store. K-tail zeros are pre-baked into the padded f16 buffers.
// Block = 256 threads = 8 waves; block tile 64(M) x 64(N); wave tile 16x32.
// ---------------------------------------------------------------------------
__global__ __launch_bounds__(256)
void wmma_gemm_kernel(const half_t* __restrict__ A,
                      const half_t* __restrict__ Bt,
                      float* __restrict__ C,
                      int M, int K, int Nout, int accFlag)
{
    __shared__ alignas(16) half_t sA[BM][LDSK];
    __shared__ alignas(16) half_t sB[BN][LDSK];

    const int tid  = threadIdx.x;
    const int lane = tid & 31;
    const int wave = tid >> 5;
    const int h    = lane >> 4;     // lane half (0/1)
    const int l15  = lane & 15;

    const int bm = blockIdx.y * BM;
    const int bn = blockIdx.x * BN;
    const int wm = bm + (wave >> 1) * 16;   // wave row base   (4 M-strips)
    const int wn = bn + (wave & 1) * 32;    // wave col base   (2 N-strips)

    // Staging: one 8-half chunk per thread per matrix (64 rows x 4 chunks = 256)
    const int sr = tid >> 2;            // 0..63
    const int sk = (tid & 3) * 8;       // 0,8,16,24
    const int ga = min(bm + sr, M - 1);      // clamped rows (see header note)
    const int gb = min(bn + sr, Nout - 1);
    const half_t* aptr = A  + (size_t)ga * K + sk;
    const half_t* bptr = Bt + (size_t)gb * K + sk;

    v8f c0 = {};
    v8f c1 = {};

    if (accFlag) {
#pragma unroll
        for (int j = 0; j < 8; ++j) {
            int row = wm + j + 8 * h;
            if (row < M) {
                int col0 = wn + l15;
                int col1 = wn + 16 + l15;
                if (col0 < Nout) c0[j] = C[(size_t)row * Nout + col0];
                if (col1 < Nout) c1[j] = C[(size_t)row * Nout + col1];
            }
        }
    }

    const int arow  = (wave >> 1) * 16 + l15;
    const int brow0 = (wave & 1) * 32 + l15;
    const int brow1 = brow0 + 16;

    for (int k0 = 0; k0 < K; k0 += BK) {
        // unconditional b128 staging loads (no exec-mask divergence)
        *(v8h*)&sA[sr][sk] = *(const v8h*)(aptr + k0);
        *(v8h*)&sB[sr][sk] = *(const v8h*)(bptr + k0);
        if (k0 + BK < K) {   // prefetch next k-tile (global_prefetch_b8)
            __builtin_prefetch(aptr + k0 + BK, 0, 1);
            __builtin_prefetch(bptr + k0 + BK, 0, 1);
        }
        __syncthreads();

        // A fragment: row = l15 (+strip), K = {8h..8h+7, 16+8h..16+8h+7}
        v8h alo = *(const v8h*)&sA[arow][8 * h];
        v8h ahi = *(const v8h*)&sA[arow][16 + 8 * h];
        v16h af;
#pragma unroll
        for (int i = 0; i < 8; ++i) { af[i] = alo[i]; af[i + 8] = ahi[i]; }

        // B fragments: col = l15 (+16 for second tile), K = {8h..8h+7, 16+8h..16+8h+7}
        v8h b0lo = *(const v8h*)&sB[brow0][8 * h];
        v8h b0hi = *(const v8h*)&sB[brow0][16 + 8 * h];
        v16h bf0;
#pragma unroll
        for (int i = 0; i < 8; ++i) { bf0[i] = b0lo[i]; bf0[i + 8] = b0hi[i]; }

        v8h b1lo = *(const v8h*)&sB[brow1][8 * h];
        v8h b1hi = *(const v8h*)&sB[brow1][16 + 8 * h];
        v16h bf1;
#pragma unroll
        for (int i = 0; i < 8; ++i) { bf1[i] = b1lo[i]; bf1[i + 8] = b1hi[i]; }

        c0 = __builtin_amdgcn_wmma_f32_16x16x32_f16(false, af, false, bf0,
                                                    (short)0, c0, false, false);
        c1 = __builtin_amdgcn_wmma_f32_16x16x32_f16(false, af, false, bf1,
                                                    (short)0, c1, false, false);
        __syncthreads();
    }

#pragma unroll
    for (int j = 0; j < 8; ++j) {
        int row = wm + j + 8 * h;
        if (row < M) {
            int col0 = wn + l15;
            int col1 = wn + 16 + l15;
            if (col0 < Nout) C[(size_t)row * Nout + col0] = c0[j];
            if (col1 < Nout) C[(size_t)row * Nout + col1] = c1[j];
        }
    }
}

// ---------------------------------------------------------------------------
// Small helper kernels
// ---------------------------------------------------------------------------
__global__ void fill_ones_kernel(float* p, int n) {
    int i = blockIdx.x * blockDim.x + threadIdx.x;
    if (i < n) p[i] = 1.0f;
}

__global__ void deg_accum_kernel(const int* __restrict__ dst, float* deg, int E) {
    int e = blockIdx.x * blockDim.x + threadIdx.x;
    if (e < E) atomicAdd(&deg[dst[e]], 1.0f);
}

__global__ void rsqrt_kernel(float* p, int n) {
    int i = blockIdx.x * blockDim.x + threadIdx.x;
    if (i < n) p[i] = rsqrtf(p[i]);
}

// f32 -> f16 straight copy
__global__ void convert_f16_kernel(const float* __restrict__ src, half_t* __restrict__ dst, size_t n) {
    size_t i = (size_t)blockIdx.x * blockDim.x + threadIdx.x;
    if (i < n) dst[i] = (half_t)src[i];
}

// f32 [rows,cols] -> f16 [rows,padcols] (zero pad tail)
__global__ void convert_pad_f16_kernel(const float* __restrict__ src, half_t* __restrict__ dst,
                                       int rows, int cols, int padcols) {
    size_t i = (size_t)blockIdx.x * blockDim.x + threadIdx.x;
    size_t total = (size_t)rows * (size_t)padcols;
    if (i >= total) return;
    int r = (int)(i / padcols);
    int c = (int)(i % padcols);
    dst[i] = (c < cols) ? (half_t)src[(size_t)r * cols + c] : (half_t)0.f;
}

// Wt[n, kpad] = (k < K) ? W[k, n] : 0   (transpose + f16 convert, pad K)
__global__ void transpose_pad_f16_kernel(const float* __restrict__ W, half_t* __restrict__ Wt,
                                         int K, int Kpad, int Nout) {
    int k = blockIdx.x * blockDim.x + threadIdx.x;
    int n = blockIdx.y;
    if (k < Kpad)
        Wt[(size_t)n * Kpad + k] = (k < K) ? (half_t)W[(size_t)k * Nout + n] : (half_t)0.f;
}

__global__ void vec_add_kernel(const float* a, const float* b, float* o, int n) {
    int i = blockIdx.x * blockDim.x + threadIdx.x;
    if (i < n) o[i] = a[i] + b[i];
}

// agg[i,f] = t[i,f] * dis[i]^2 + (b ? b[f] : 0)      block = 256, grid = N
__global__ void prop_init_kernel(const float* __restrict__ t, const float* __restrict__ dis,
                                 const float* __restrict__ b, float* __restrict__ agg) {
    int i = blockIdx.x;
    int f = threadIdx.x;
    float s = dis[i]; s *= s;
    size_t off = (size_t)i * NHID + f;
    agg[off] = t[off] * s + (b ? b[f] : 0.f);
}

// agg[dst,f] += t[src,f] * dis[src]*dis[dst]         block = 256, grid = E
__global__ void prop_edges_kernel(const float* __restrict__ t, const float* __restrict__ dis,
                                  const int* __restrict__ src, const int* __restrict__ dstv,
                                  float* __restrict__ agg) {
    int e = blockIdx.x;
    int f = threadIdx.x;
    int s = src[e], d = dstv[e];
    float c = dis[s] * dis[d];
    atomicAdd(&agg[(size_t)d * NHID + f], t[(size_t)s * NHID + f] * c);
}

// h = relu(agg); write f32 and f16 copies
__global__ void relu_store_kernel(const float* __restrict__ agg, float* __restrict__ h32,
                                  half_t* __restrict__ h16, size_t n) {
    size_t i = (size_t)blockIdx.x * blockDim.x + threadIdx.x;
    if (i < n) {
        float v = fmaxf(agg[i], 0.f);
        h32[i] = v;
        h16[i] = (half_t)v;
    }
}

__device__ __forceinline__ float sigf(float x) { return 1.f / (1.f + __expf(-x)); }

// gates G[N,1024] in i,f,g,o order; bsum = bih+bhh; c_in nullable (zero state)
__global__ void lstm_pointwise_kernel(const float* __restrict__ G, const float* __restrict__ bsum,
                                      const float* __restrict__ c_in,
                                      float* __restrict__ h32, half_t* __restrict__ h16,
                                      float* __restrict__ c_out) {
    int i = blockIdx.x;
    int f = threadIdx.x;
    size_t gb = (size_t)i * 4 * NHID;
    float gi = G[gb + f]            + bsum[f];
    float gf = G[gb + NHID + f]     + bsum[NHID + f];
    float gg = G[gb + 2 * NHID + f] + bsum[2 * NHID + f];
    float go = G[gb + 3 * NHID + f] + bsum[3 * NHID + f];
    size_t off = (size_t)i * NHID + f;
    float ci = c_in ? c_in[off] : 0.f;
    float c  = sigf(gf) * ci + sigf(gi) * tanhf(gg);
    float hv = sigf(go) * tanhf(c);
    h32[off] = hv;
    if (h16)  h16[off]  = (half_t)hv;
    if (c_out) c_out[off] = c;
}

// JK attention + combine: block = 256, grid = N
__global__ void jk_attention_kernel(const float* __restrict__ hf1, const float* __restrict__ hb1,
                                    const float* __restrict__ hf2, const float* __restrict__ hb2,
                                    const float* __restrict__ Watt, const float* __restrict__ batt,
                                    const float* __restrict__ h1, const float* __restrict__ h2,
                                    float* __restrict__ xjk) {
    __shared__ float red[256];
    int i = blockIdx.x;
    int f = threadIdx.x;
    size_t off = (size_t)i * NHID + f;
    float waf = Watt[f], wab = Watt[NHID + f];
    float p1 = hf1[off] * waf + hb1[off] * wab;
    float p2 = hf2[off] * waf + hb2[off] * wab;

    red[f] = p1; __syncthreads();
    for (int s = 128; s > 0; s >>= 1) { if (f < s) red[f] += red[f + s]; __syncthreads(); }
    float a1 = red[0] + batt[0];
    __syncthreads();
    red[f] = p2; __syncthreads();
    for (int s = 128; s > 0; s >>= 1) { if (f < s) red[f] += red[f + s]; __syncthreads(); }
    float a2 = red[0] + batt[0];

    float m  = fmaxf(a1, a2);
    float e1 = __expf(a1 - m), e2 = __expf(a2 - m);
    float inv = 1.f / (e1 + e2);
    xjk[off] = (e1 * inv) * h1[off] + (e2 * inv) * h2[off];
}

// out = log_softmax(Y + bl) over 40 classes, one thread per node
__global__ void log_softmax_kernel(const float* __restrict__ Y, const float* __restrict__ bl,
                                   float* __restrict__ out, int n) {
    int i = blockIdx.x * blockDim.x + threadIdx.x;
    if (i >= n) return;
    float v[NCLS];
    float m = -INFINITY;
#pragma unroll
    for (int j = 0; j < NCLS; ++j) { v[j] = Y[(size_t)i * NCLS + j] + bl[j]; m = fmaxf(m, v[j]); }
    float s = 0.f;
#pragma unroll
    for (int j = 0; j < NCLS; ++j) s += __expf(v[j] - m);
    float lse = m + __logf(s);
#pragma unroll
    for (int j = 0; j < NCLS; ++j) out[(size_t)i * NCLS + j] = v[j] - lse;
}

// ---------------------------------------------------------------------------
// Host side
// ---------------------------------------------------------------------------
extern "C" void kernel_launch(void* const* d_in, const int* in_sizes, int n_in,
                              void* d_out, int out_size, void* d_ws, size_t ws_size,
                              hipStream_t stream) {
    const float* x     = (const float*)d_in[0];
    const int*   ei    = (const int*)  d_in[1];
    const float* W1    = (const float*)d_in[2];
    const float* b1    = (const float*)d_in[3];
    const float* W2    = (const float*)d_in[4];
    const float* b2    = (const float*)d_in[5];
    const float* WihF  = (const float*)d_in[6];
    const float* WhhF  = (const float*)d_in[7];
    const float* bihF  = (const float*)d_in[8];
    const float* bhhF  = (const float*)d_in[9];
    const float* WihB  = (const float*)d_in[10];
    const float* WhhB  = (const float*)d_in[11];
    const float* bihB  = (const float*)d_in[12];
    const float* bhhB  = (const float*)d_in[13];
    const float* Watt  = (const float*)d_in[14];
    const float* batt  = (const float*)d_in[15];
    const float* Wl    = (const float*)d_in[16];
    const float* bl    = (const float*)d_in[17];
    float* out = (float*)d_out;

    const int N = in_sizes[0] / NFEAT;
    const int E = in_sizes[1] / 2;
    const int* esrc = ei;
    const int* edst = ei + E;

    // --- workspace carving ---
    char* wp = (char*)d_ws;
    auto alloc = [&](size_t bytes) -> void* {
        void* p = (void*)wp;
        wp += (bytes + 255) & ~(size_t)255;
        return p;
    };
    const size_t NH  = (size_t)N * NHID;
    float*  dis   = (float*) alloc((size_t)N * 4);
    half_t* x16   = (half_t*)alloc((size_t)N * KPAD * 2);
    half_t* W1t   = (half_t*)alloc((size_t)NHID * KPAD * 2);
    half_t* W2t   = (half_t*)alloc((size_t)NHID * NHID * 2);
    half_t* Wlt   = (half_t*)alloc((size_t)NCLS * NHID * 2);
    half_t* WihF16= (half_t*)alloc((size_t)4 * NHID * NHID * 2);
    half_t* WhhF16= (half_t*)alloc((size_t)4 * NHID * NHID * 2);
    half_t* WihB16= (half_t*)alloc((size_t)4 * NHID * NHID * 2);
    half_t* WhhB16= (half_t*)alloc((size_t)4 * NHID * NHID * 2);
    float*  bsumF = (float*) alloc((size_t)4 * NHID * 4);
    float*  bsumB = (float*) alloc((size_t)4 * NHID * 4);
    float*  T     = (float*) alloc(NH * 4);   // linear output scratch
    float*  AGG   = (float*) alloc(NH * 4);   // aggregation scratch
    float*  H1    = (float*) alloc(NH * 4);
    float*  H2    = (float*) alloc(NH * 4);
    float*  HF1   = (float*) alloc(NH * 4);
    float*  HF2   = (float*) alloc(NH * 4);
    float*  HB1   = (float*) alloc(NH * 4);
    float*  HB2   = (float*) alloc(NH * 4);
    float*  CBUF  = (float*) alloc(NH * 4);   // cf1 then cb2
    float*  XJK   = (float*) alloc(NH * 4);
    half_t* h1_16 = (half_t*)alloc(NH * 2);
    half_t* h2_16 = (half_t*)alloc(NH * 2);
    half_t* hf1_16= (half_t*)alloc(NH * 2);
    half_t* hb2_16= (half_t*)alloc(NH * 2);
    half_t* xp16  = (half_t*)alloc(NH * 2);
    float*  G     = (float*) alloc((size_t)N * 4 * NHID * 4);
    float*  Y     = (float*) alloc((size_t)N * NCLS * 4);

    auto cdiv = [](size_t a, size_t b) { return (unsigned)((a + b - 1) / b); };
    auto gemm = [&](const half_t* A, const half_t* Bt, float* C, int M, int K, int Nout, int acc) {
        dim3 g((Nout + BN - 1) / BN, (M + BM - 1) / BM);
        wmma_gemm_kernel<<<g, 256, 0, stream>>>(A, Bt, C, M, K, Nout, acc);
    };

    // 1. degree -> dis = rsqrt(deg + 1)
    fill_ones_kernel<<<cdiv(N, 256), 256, 0, stream>>>(dis, N);
    deg_accum_kernel<<<cdiv(E, 256), 256, 0, stream>>>(edst, dis, E);
    rsqrt_kernel<<<cdiv(N, 256), 256, 0, stream>>>(dis, N);

    // 2. operand prep (f16 converts / transposes, bias sums)
    convert_pad_f16_kernel<<<cdiv((size_t)N * KPAD, 256), 256, 0, stream>>>(x, x16, N, NFEAT, KPAD);
    {
        dim3 g1(cdiv(KPAD, 256), NHID);
        transpose_pad_f16_kernel<<<g1, 256, 0, stream>>>(W1, W1t, NFEAT, KPAD, NHID);
        dim3 g2(1, NHID);
        transpose_pad_f16_kernel<<<g2, 256, 0, stream>>>(W2, W2t, NHID, NHID, NHID);
        dim3 g3(1, NCLS);
        transpose_pad_f16_kernel<<<g3, 256, 0, stream>>>(Wl, Wlt, NHID, NHID, NCLS);
    }
    const size_t wsz = (size_t)4 * NHID * NHID;
    convert_f16_kernel<<<cdiv(wsz, 256), 256, 0, stream>>>(WihF, WihF16, wsz);
    convert_f16_kernel<<<cdiv(wsz, 256), 256, 0, stream>>>(WhhF, WhhF16, wsz);
    convert_f16_kernel<<<cdiv(wsz, 256), 256, 0, stream>>>(WihB, WihB16, wsz);
    convert_f16_kernel<<<cdiv(wsz, 256), 256, 0, stream>>>(WhhB, WhhB16, wsz);
    vec_add_kernel<<<4, 256, 0, stream>>>(bihF, bhhF, bsumF, 4 * NHID);
    vec_add_kernel<<<4, 256, 0, stream>>>(bihB, bhhB, bsumB, 4 * NHID);

    // 3. layer 1: h1 = relu(gcn_prop(x@W1) + b1)
    gemm(x16, W1t, T, N, KPAD, NHID, 0);
    prop_init_kernel<<<N, NHID, 0, stream>>>(T, dis, b1, AGG);
    prop_edges_kernel<<<E, NHID, 0, stream>>>(T, dis, esrc, edst, AGG);
    relu_store_kernel<<<cdiv(NH, 256), 256, 0, stream>>>(AGG, H1, h1_16, NH);

    // 4. layer 2: h2 = relu(gcn_prop(h1@W2) + b2)
    gemm(h1_16, W2t, T, N, NHID, NHID, 0);
    prop_init_kernel<<<N, NHID, 0, stream>>>(T, dis, b2, AGG);
    prop_edges_kernel<<<E, NHID, 0, stream>>>(T, dis, esrc, edst, AGG);
    relu_store_kernel<<<cdiv(NH, 256), 256, 0, stream>>>(AGG, H2, h2_16, NH);

    // 5. BiLSTM JumpingKnowledge over the length-2 sequence [h1, h2]
    // forward t=1: state 0
    gemm(h1_16, WihF16, G, N, NHID, 4 * NHID, 0);
    lstm_pointwise_kernel<<<N, NHID, 0, stream>>>(G, bsumF, nullptr, HF1, hf1_16, CBUF);
    // forward t=2: state (hf1, cf1)
    gemm(h2_16, WihF16, G, N, NHID, 4 * NHID, 0);
    gemm(hf1_16, WhhF16, G, N, NHID, 4 * NHID, 1);
    lstm_pointwise_kernel<<<N, NHID, 0, stream>>>(G, bsumF, CBUF, HF2, nullptr, nullptr);
    // backward t=2: state 0
    gemm(h2_16, WihB16, G, N, NHID, 4 * NHID, 0);
    lstm_pointwise_kernel<<<N, NHID, 0, stream>>>(G, bsumB, nullptr, HB2, hb2_16, CBUF);
    // backward t=1: state (hb2, cb2)
    gemm(h1_16, WihB16, G, N, NHID, 4 * NHID, 0);
    gemm(hb2_16, WhhB16, G, N, NHID, 4 * NHID, 1);
    lstm_pointwise_kernel<<<N, NHID, 0, stream>>>(G, bsumB, CBUF, HB1, nullptr, nullptr);

    // 6. attention combine -> xjk
    jk_attention_kernel<<<N, NHID, 0, stream>>>(HF1, HB1, HF2, HB2, Watt, batt, H1, H2, XJK);

    // 7. APPNP(K=1, alpha=0): xp = gcn_prop(xjk)   (reuse T as agg buffer)
    prop_init_kernel<<<N, NHID, 0, stream>>>(XJK, dis, nullptr, T);
    prop_edges_kernel<<<E, NHID, 0, stream>>>(XJK, dis, esrc, edst, T);
    convert_f16_kernel<<<cdiv(NH, 256), 256, 0, stream>>>(T, xp16, NH);

    // 8. classifier + log_softmax
    gemm(xp16, Wlt, Y, N, NHID, NCLS, 0);
    log_softmax_kernel<<<cdiv(N, 256), 256, 0, stream>>>(Y, bl, out, N);

    (void)n_in; (void)out_size; (void)ws_size;
}
